// BiLSTMCRF_17179869706
// MI455X (gfx1250) — compile-verified, matrix operand order checked by reading, not run on hardware
//
#include <hip/hip_runtime.h>
#include <hip/hip_bf16.h>

typedef __attribute__((ext_vector_type(16))) _Float16 v16h;
typedef __attribute__((ext_vector_type(8)))  _Float16 v8h;
typedef __attribute__((ext_vector_type(8)))  float    v8f;

constexpr int SEQ_T = 512;
constexpr int BATCH = 32;
constexpr int EDIM  = 128;
constexpr int HDIM  = 256;      // hidden per direction
constexpr int GDIM  = 4 * HDIM; // 1024 gate width
constexpr int TBROWS = SEQ_T * BATCH; // 16384

// LDS weight cache for the recurrent scan: waves 0..3 read Whh from LDS.
constexpr int WCACHE_WAVES    = 4;
constexpr int WHH_LDS_HALVES  = WCACHE_WAVES * 4 * 32 * HDIM; // 131072 (256 KB)
constexpr int HST_HALVES      = BATCH * HDIM;                 // 8192   (16 KB)
constexpr size_t REC_LDS_BYTES = (size_t)(WHH_LDS_HALVES + HST_HALVES) * 2; // 272 KB

// ---------------------------------------------------------------------------
// helpers
// ---------------------------------------------------------------------------
__device__ __forceinline__ float sigf(float x) { return 1.0f / (1.0f + __expf(-x)); }

// 16x32 f16 fragment: lane holds one row (row_lane) of a row-major matrix.
// ISA 7.12.2: lanes 0-15 hold K=[kk,kk+8)+[kk+16,kk+24); lanes 16-31 the other halves.
__device__ __forceinline__ v16h load_frag(const _Float16* __restrict__ base,
                                          int stride, int row_lane, int kk, int hi) {
  const _Float16* p = base + (size_t)row_lane * stride + kk + hi * 8;
  union { v16h v; v8h h[2]; } u;
  u.h[0] = *(const v8h*)(p);
  u.h[1] = *(const v8h*)(p + 16);
  return u.v;
}

__device__ __forceinline__ v16h load_frag_lds(const _Float16* base,
                                              int stride, int row_lane, int kk, int hi) {
  const _Float16* p = base + row_lane * stride + kk + hi * 8;
  union { v16h v; v8h h[2]; } u;
  u.h[0] = *(const v8h*)(p);
  u.h[1] = *(const v8h*)(p + 16);
  return u.v;
}

#define WMMA_F16(a, b, c) __builtin_amdgcn_wmma_f32_16x16x32_f16( \
    false, (a), false, (b), (short)0, (c), false, false)

// ---------------------------------------------------------------------------
// 1. f32 -> f16 weight conversion
// ---------------------------------------------------------------------------
__global__ void f32_to_f16_kernel(const float* __restrict__ src,
                                  _Float16* __restrict__ dst, int n) {
  int i = blockIdx.x * blockDim.x + threadIdx.x;
  if (i < n) dst[i] = (_Float16)src[i];
}

// ---------------------------------------------------------------------------
// 2. embedding gather: x_emb[(t*B+b)*E+e] = f16(emb[x[b,t]*E+e])
// ---------------------------------------------------------------------------
__global__ void embed_kernel(const int* __restrict__ x,
                             const float* __restrict__ emb,
                             _Float16* __restrict__ x_emb) {
  size_t idx = (size_t)blockIdx.x * blockDim.x + threadIdx.x;
  if (idx >= (size_t)TBROWS * EDIM) return;
  int e  = (int)(idx % EDIM);
  int tb = (int)(idx / EDIM);          // tb = t*B + b
  int b  = tb % BATCH;
  int t  = tb / BATCH;
  int tok = x[b * SEQ_T + t];
  x_emb[idx] = (_Float16)emb[(size_t)tok * EDIM + e];
}

// ---------------------------------------------------------------------------
// 3a. Parallel input projection (off the recurrent critical path):
//     Gt[n][tb] = f16( bias[n] + sum_k x_in[tb,k] * Wih[n,k] ), stored
//     transposed so the scan's accumulator init is one b128 load per tile.
// ---------------------------------------------------------------------------
__global__ __launch_bounds__(256) void input_proj_kernel(
    const _Float16* __restrict__ x_in, int in_dim,
    const _Float16* __restrict__ Wih, const float* __restrict__ bias,
    _Float16* __restrict__ Gt) {
  const int lane = threadIdx.x & 31;
  const int w    = threadIdx.x >> 5;
  const int hi   = lane >> 4;
  const int nl   = lane & 15;
  const int gw   = blockIdx.x * 8 + w;          // wave job id
  const int mtile  = gw >> 4;                   // 0..1023 (16 rows each)
  const int nbase0 = (gw & 15) * 64;            // 0..960

  const _Float16* xrow = x_in + (size_t)(mtile * 16) * in_dim;
  v8f acc[4] = {};
  for (int kk = 0; kk < in_dim; kk += 32) {
    v16h a = load_frag(xrow, in_dim, nl, kk, hi);
    v16h bfr[4];
#pragma unroll
    for (int u = 0; u < 4; ++u)
      bfr[u] = load_frag(Wih + (size_t)(nbase0 + u * 16) * in_dim, in_dim, nl, kk, hi);
#pragma unroll
    for (int u = 0; u < 4; ++u)
      acc[u] = WMMA_F16(a, bfr[u], acc[u]);
  }
#pragma unroll
  for (int u = 0; u < 4; ++u) {
    const int n = nbase0 + u * 16 + nl;
    const float bn = bias[n];
    v8h out;
#pragma unroll
    for (int r = 0; r < 8; ++r) out[r] = (_Float16)(acc[u][r] + bn);
    *(v8h*)&Gt[(size_t)n * TBROWS + mtile * 16 + 8 * hi] = out;
  }
}

// ---------------------------------------------------------------------------
// 3b. Persistent recurrent-only LSTM scan. grid.x = 2 (dir), block = 256.
//     - Accumulators init'd from precomputed Gt (bias folded in).
//     - Whh rows for waves 0..3 cached in 256 KB of LDS (one-time copy);
//       waves 4..7 stream theirs from L2. 272 KB dynamic LDS total.
//     - K-loop double-buffers fragments so chunk k+1 loads overlap chunk k's
//       16 back-to-back WMMAs.
//     - Next timestep's Gt lines prefetched (global_prefetch_b8).
// ---------------------------------------------------------------------------
__global__ __launch_bounds__(256) void lstm_rec_kernel(
    const _Float16* __restrict__ Gt_f, const _Float16* __restrict__ Gt_b,
    const _Float16* __restrict__ Whh_f, const _Float16* __restrict__ Whh_b,
    _Float16* __restrict__ h_out /* (T*B, 512), dir half-offset */) {
  const int dir = blockIdx.x;
  const _Float16* __restrict__ Gt  = dir ? Gt_b  : Gt_f;
  const _Float16* __restrict__ Whh = dir ? Whh_b : Whh_f;

  extern __shared__ _Float16 smem[];
  _Float16* whh_lds = smem;                 // packed [(wv*4+gi)*32 + jj][k]
  _Float16* h_st    = smem + WHH_LDS_HALVES;

  const int tid  = threadIdx.x;
  const int lane = tid & 31;
  const int w    = tid >> 5;
  const int hi   = lane >> 4;
  const int nl   = lane & 15;

  // one-time: stage Whh rows of waves 0..3 into LDS (b128 copies)
  for (int v = tid; v < WHH_LDS_HALVES / 8; v += 256) {
    const int hidx = v * 8;
    const int k   = hidx & (HDIM - 1);
    const int row = hidx >> 8;              // HDIM == 256
    const int jj = row & 31;
    const int gi = (row >> 5) & 3;
    const int wv = row >> 7;
    const int grow = gi * HDIM + wv * 32 + jj;
    *(v8h*)&whh_lds[hidx] = *(const v8h*)&Whh[(size_t)grow * HDIM + k];
  }
  for (int i = tid; i < HST_HALVES; i += 256) h_st[i] = (_Float16)0.0f;
  __syncthreads();

  v8f c[2][2] = {}; // cell state per [mt][jt]

  auto load_chunk = [&](int kk, v16h& a0, v16h& a1, v16h* bfr) {
    a0 = load_frag_lds(h_st, HDIM, nl, kk, hi);
    a1 = load_frag_lds(h_st + 16 * HDIM, HDIM, nl, kk, hi);
    if (w < WCACHE_WAVES) { // wave-uniform branch: EXEC stays all-ones
#pragma unroll
      for (int u = 0; u < 8; ++u) {
        const int gi = u >> 1, jt = u & 1;
        bfr[u] = load_frag_lds(whh_lds + ((w * 4 + gi) * 32) * HDIM,
                               HDIM, jt * 16 + nl, kk, hi);
      }
    } else {
#pragma unroll
      for (int u = 0; u < 8; ++u) {
        const int gi = u >> 1, jt = u & 1;
        bfr[u] = load_frag(Whh + (size_t)(gi * HDIM + w * 32 + jt * 16) * HDIM,
                           HDIM, nl, kk, hi);
      }
    }
  };

  for (int step = 0; step < SEQ_T; ++step) {
    const int t = dir ? (SEQ_T - 1 - step) : step;

    // init accumulators from precomputed input projection (+bias)
    v8f acc[2][2][4];
#pragma unroll
    for (int mt = 0; mt < 2; ++mt) {
#pragma unroll
      for (int jt = 0; jt < 2; ++jt) {
#pragma unroll
        for (int gi = 0; gi < 4; ++gi) {
          const int n = gi * HDIM + w * 32 + jt * 16 + nl;
          const v8h g8 = *(const v8h*)&Gt[(size_t)n * TBROWS + t * BATCH + mt * 16 + 8 * hi];
#pragma unroll
          for (int r = 0; r < 8; ++r) acc[mt][jt][gi][r] = (float)g8[r];
        }
      }
    }

    // prefetch next step's Gt lines toward this WGP
    if (step + 1 < SEQ_T) {
      const int tn = dir ? (SEQ_T - 2 - step) : (step + 1);
#pragma unroll
      for (int mt = 0; mt < 2; ++mt) {
#pragma unroll
        for (int jt = 0; jt < 2; ++jt) {
#pragma unroll
          for (int gi = 0; gi < 4; ++gi) {
            const int n = gi * HDIM + w * 32 + jt * 16 + nl;
            __builtin_prefetch(&Gt[(size_t)n * TBROWS + tn * BATCH + mt * 16 + 8 * hi], 0, 3);
          }
        }
      }
    }

    // recurrent projection: K = 256, double-buffered fragment pipeline
    {
      v16h a0[2], a1[2], bfr[2][8];
      load_chunk(0, a0[0], a1[0], bfr[0]);
#pragma unroll
      for (int ki = 0; ki < 8; ++ki) {
        const int cur = ki & 1;
        if (ki < 7) load_chunk((ki + 1) * 32, a0[cur ^ 1], a1[cur ^ 1], bfr[cur ^ 1]);
#pragma unroll
        for (int u = 0; u < 8; ++u) {
          const int jt = u & 1, gi = u >> 1;
          acc[0][jt][gi] = WMMA_F16(a0[cur], bfr[cur][u], acc[0][jt][gi]);
          acc[1][jt][gi] = WMMA_F16(a1[cur], bfr[cur][u], acc[1][jt][gi]);
        }
      }
    }

    __syncthreads(); // all h_st reads done

    // LSTM cell: i/f/g/o for one (b,j) share a lane/VGPR slot
#pragma unroll
    for (int mt = 0; mt < 2; ++mt) {
#pragma unroll
      for (int jt = 0; jt < 2; ++jt) {
#pragma unroll
        for (int r = 0; r < 8; ++r) {
          float iv = sigf(acc[mt][jt][0][r]);
          float fv = sigf(acc[mt][jt][1][r]);
          float gv = tanhf(acc[mt][jt][2][r]);
          float ov = sigf(acc[mt][jt][3][r]);
          float cv = fv * c[mt][jt][r] + iv * gv;
          c[mt][jt][r] = cv;
          float hv = ov * tanhf(cv);
          const int b = mt * 16 + r + 8 * hi;
          const int j = w * 32 + jt * 16 + nl;
          _Float16 hh = (_Float16)hv;
          h_st[b * HDIM + j] = hh;
          h_out[(size_t)(t * BATCH + b) * (2 * HDIM) + dir * HDIM + j] = hh;
        }
      }
    }
    __syncthreads();
  }
}

// ---------------------------------------------------------------------------
// 3c. Fused fallback (small workspace): input + recurrent GEMM per step.
// ---------------------------------------------------------------------------
__global__ __launch_bounds__(256) void lstm_layer_fused_kernel(
    const _Float16* __restrict__ x_in, int in_dim,
    const _Float16* __restrict__ Wih_f, const _Float16* __restrict__ Whh_f,
    const float* __restrict__ bias_f,
    const _Float16* __restrict__ Wih_b, const _Float16* __restrict__ Whh_b,
    const float* __restrict__ bias_b,
    _Float16* __restrict__ h_out) {
  const int dir = blockIdx.x;
  const _Float16* __restrict__ Wih  = dir ? Wih_b  : Wih_f;
  const _Float16* __restrict__ Whh  = dir ? Whh_b  : Whh_f;
  const float*    __restrict__ bias = dir ? bias_b : bias_f;

  __shared__ _Float16 h_st[BATCH * HDIM];

  const int tid  = threadIdx.x;
  const int lane = tid & 31;
  const int w    = tid >> 5;
  const int hi   = lane >> 4;
  const int nl   = lane & 15;

  for (int i = tid; i < BATCH * HDIM; i += 256) h_st[i] = (_Float16)0.0f;
  __syncthreads();

  float bv[4][2];
#pragma unroll
  for (int gi = 0; gi < 4; ++gi)
#pragma unroll
    for (int jt = 0; jt < 2; ++jt)
      bv[gi][jt] = bias[gi * HDIM + w * 32 + jt * 16 + nl];

  v8f c[2][2] = {};

  for (int step = 0; step < SEQ_T; ++step) {
    const int t = dir ? (SEQ_T - 1 - step) : step;
    v8f acc[2][2][4] = {};

    const _Float16* xrow = x_in + (size_t)(t * BATCH) * in_dim;
    for (int kk = 0; kk < in_dim; kk += 32) {
      v16h a0 = load_frag(xrow, in_dim, nl, kk, hi);
      v16h a1 = load_frag(xrow + (size_t)16 * in_dim, in_dim, nl, kk, hi);
#pragma unroll
      for (int half = 0; half < 2; ++half) {
        v16h bfr[4];
#pragma unroll
        for (int u4 = 0; u4 < 4; ++u4) {
          const int u = half * 4 + u4;
          const int nbase = (u >> 1) * HDIM + w * 32 + (u & 1) * 16;
          bfr[u4] = load_frag(Wih + (size_t)nbase * in_dim, in_dim, nl, kk, hi);
        }
#pragma unroll
        for (int u4 = 0; u4 < 4; ++u4) {
          const int u = half * 4 + u4;
          const int jt = u & 1, gi = u >> 1;
          acc[0][jt][gi] = WMMA_F16(a0, bfr[u4], acc[0][jt][gi]);
          acc[1][jt][gi] = WMMA_F16(a1, bfr[u4], acc[1][jt][gi]);
        }
      }
    }
#pragma unroll
    for (int kk = 0; kk < HDIM; kk += 32) {
      v16h a0 = load_frag_lds(&h_st[0], HDIM, nl, kk, hi);
      v16h a1 = load_frag_lds(&h_st[16 * HDIM], HDIM, nl, kk, hi);
#pragma unroll
      for (int half = 0; half < 2; ++half) {
        v16h bfr[4];
#pragma unroll
        for (int u4 = 0; u4 < 4; ++u4) {
          const int u = half * 4 + u4;
          const int nbase = (u >> 1) * HDIM + w * 32 + (u & 1) * 16;
          bfr[u4] = load_frag(Whh + (size_t)nbase * HDIM, HDIM, nl, kk, hi);
        }
#pragma unroll
        for (int u4 = 0; u4 < 4; ++u4) {
          const int u = half * 4 + u4;
          const int jt = u & 1, gi = u >> 1;
          acc[0][jt][gi] = WMMA_F16(a0, bfr[u4], acc[0][jt][gi]);
          acc[1][jt][gi] = WMMA_F16(a1, bfr[u4], acc[1][jt][gi]);
        }
      }
    }

    __syncthreads();

#pragma unroll
    for (int mt = 0; mt < 2; ++mt) {
#pragma unroll
      for (int jt = 0; jt < 2; ++jt) {
#pragma unroll
        for (int r = 0; r < 8; ++r) {
          float iv = sigf(acc[mt][jt][0][r] + bv[0][jt]);
          float fv = sigf(acc[mt][jt][1][r] + bv[1][jt]);
          float gv = tanhf(acc[mt][jt][2][r] + bv[2][jt]);
          float ov = sigf(acc[mt][jt][3][r] + bv[3][jt]);
          float cv = fv * c[mt][jt][r] + iv * gv;
          c[mt][jt][r] = cv;
          float hv = ov * tanhf(cv);
          const int b = mt * 16 + r + 8 * hi;
          const int j = w * 32 + jt * 16 + nl;
          _Float16 hh = (_Float16)hv;
          h_st[b * HDIM + j] = hh;
          h_out[(size_t)(t * BATCH + b) * (2 * HDIM) + dir * HDIM + j] = hh;
        }
      }
    }
    __syncthreads();
  }
}

// ---------------------------------------------------------------------------
// 4. emission projection (L=2): one wave per (t,b), shuffle reduction.
// ---------------------------------------------------------------------------
__global__ __launch_bounds__(256) void emission_kernel(
    const _Float16* __restrict__ h2, const float* __restrict__ W_out,
    const float* __restrict__ b_out, float* __restrict__ em) {
  const int gw = blockIdx.x * 8 + (threadIdx.x >> 5);
  const int lane = threadIdx.x & 31;
  if (gw >= TBROWS) return;
  const _Float16* hrow = h2 + (size_t)gw * (2 * HDIM);
  float a0 = 0.f, a1 = 0.f;
  for (int k = lane; k < 2 * HDIM; k += 32) {
    float hv = (float)hrow[k];
    a0 += hv * W_out[k];
    a1 += hv * W_out[2 * HDIM + k];
  }
  for (int off = 16; off; off >>= 1) {
    a0 += __shfl_down(a0, off, 32);
    a1 += __shfl_down(a1, off, 32);
  }
  if (lane == 0) {
    em[gw * 2 + 0] = a0 + b_out[0];
    em[gw * 2 + 1] = a1 + b_out[1];
  }
}

// ---------------------------------------------------------------------------
// 5. CRF forward + gold score + mean loss. One block, 32 threads, lane = batch.
// ---------------------------------------------------------------------------
__global__ void crf_kernel(const float* __restrict__ em,
                           const int* __restrict__ labels,
                           const unsigned char* __restrict__ mask,
                           const float* __restrict__ trans,
                           const float* __restrict__ startt,
                           const float* __restrict__ endt,
                           float* __restrict__ out) {
  const int b = threadIdx.x;
  const float t00 = trans[0], t01 = trans[1], t10 = trans[2], t11 = trans[3];

  float s0 = startt[0] + em[(0 * BATCH + b) * 2 + 0];
  float s1 = startt[1] + em[(0 * BATCH + b) * 2 + 1];

  int lab_prev = labels[b * SEQ_T + 0];
  float gold = startt[lab_prev] + em[(0 * BATCH + b) * 2 + lab_prev];
  int cnt = (mask[b * SEQ_T + 0] != 0) ? 1 : 0;

  for (int t = 1; t < SEQ_T; ++t) {
    const float e0 = em[((size_t)t * BATCH + b) * 2 + 0];
    const float e1 = em[((size_t)t * BATCH + b) * 2 + 1];
    float m0 = fmaxf(s0 + t00, s1 + t10);
    float n0 = m0 + __logf(__expf(s0 + t00 - m0) + __expf(s1 + t10 - m0)) + e0;
    float m1 = fmaxf(s0 + t01, s1 + t11);
    float n1 = m1 + __logf(__expf(s0 + t01 - m1) + __expf(s1 + t11 - m1)) + e1;
    const bool mk = mask[b * SEQ_T + t] != 0;
    s0 = mk ? n0 : s0;
    s1 = mk ? n1 : s1;

    const int lab = labels[b * SEQ_T + t];
    const float tr = trans[lab_prev * 2 + lab];
    const float el = mk ? em[((size_t)t * BATCH + b) * 2 + lab] : 0.f;
    gold += mk ? (tr + el) : 0.f;
    if (mk) cnt++;
    lab_prev = lab;
  }
  const int last = labels[b * SEQ_T + (cnt - 1)];
  gold += endt[last];

  float mz = fmaxf(s0 + endt[0], s1 + endt[1]);
  float log_z = mz + __logf(__expf(s0 + endt[0] - mz) + __expf(s1 + endt[1] - mz));

  float loss = log_z - gold;
  for (int off = 16; off; off >>= 1) loss += __shfl_down(loss, off, 32);
  if (b == 0) out[0] = loss / (float)BATCH;
}

// ---------------------------------------------------------------------------
// launch
// ---------------------------------------------------------------------------
extern "C" void kernel_launch(void* const* d_in, const int* in_sizes, int n_in,
                              void* d_out, int out_size, void* d_ws, size_t ws_size,
                              hipStream_t stream) {
  const int*   x        = (const int*)d_in[0];
  const int*   labels   = (const int*)d_in[2];
  const unsigned char* mask = (const unsigned char*)d_in[3];
  const float* emb      = (const float*)d_in[4];
  const float* Wih0f    = (const float*)d_in[5];
  const float* Whh0f    = (const float*)d_in[6];
  const float* b0f      = (const float*)d_in[7];
  const float* Wih0b    = (const float*)d_in[8];
  const float* Whh0b    = (const float*)d_in[9];
  const float* b0b      = (const float*)d_in[10];
  const float* Wih1f    = (const float*)d_in[11];
  const float* Whh1f    = (const float*)d_in[12];
  const float* b1f      = (const float*)d_in[13];
  const float* Wih1b    = (const float*)d_in[14];
  const float* Whh1b    = (const float*)d_in[15];
  const float* b1b      = (const float*)d_in[16];
  const float* W_out    = (const float*)d_in[17];
  const float* b_out    = (const float*)d_in[18];
  const float* trans    = (const float*)d_in[19];
  const float* startt   = (const float*)d_in[20];
  const float* endt     = (const float*)d_in[21];

  char* ws = (char*)d_ws;
  _Float16* x_emb = (_Float16*)(ws);                              // 4 MB
  _Float16* h1    = (_Float16*)(ws + ((size_t)4  << 20));         // 16 MB
  _Float16* h2    = (_Float16*)(ws + ((size_t)20 << 20));         // 16 MB
  float*    em    = (float*)   (ws + ((size_t)36 << 20));         // 128 KB
  _Float16* wbuf  = (_Float16*)(ws + ((size_t)37 << 20));         // ~4.5 MB
  _Float16* Gf    = (_Float16*)(ws + ((size_t)42 << 20));         // 32 MB
  _Float16* Gb    = (_Float16*)(ws + ((size_t)74 << 20));         // 32 MB

  const bool useG = ws_size >= ((size_t)107 << 20);

  _Float16* hWih0f = wbuf;
  _Float16* hWih0b = hWih0f + GDIM * EDIM;
  _Float16* hWhh0f = hWih0b + GDIM * EDIM;
  _Float16* hWhh0b = hWhh0f + GDIM * HDIM;
  _Float16* hWih1f = hWhh0b + GDIM * HDIM;
  _Float16* hWih1b = hWih1f + GDIM * 2 * HDIM;
  _Float16* hWhh1f = hWih1b + GDIM * 2 * HDIM;
  _Float16* hWhh1b = hWhh1f + GDIM * HDIM;

  auto cvt = [&](const float* s, _Float16* d, int n) {
    f32_to_f16_kernel<<<(n + 255) / 256, 256, 0, stream>>>(s, d, n);
  };
  cvt(Wih0f, hWih0f, GDIM * EDIM);
  cvt(Wih0b, hWih0b, GDIM * EDIM);
  cvt(Whh0f, hWhh0f, GDIM * HDIM);
  cvt(Whh0b, hWhh0b, GDIM * HDIM);
  cvt(Wih1f, hWih1f, GDIM * 2 * HDIM);
  cvt(Wih1b, hWih1b, GDIM * 2 * HDIM);
  cvt(Whh1f, hWhh1f, GDIM * HDIM);
  cvt(Whh1b, hWhh1b, GDIM * HDIM);

  {
    const size_t n = (size_t)TBROWS * EDIM;
    embed_kernel<<<(int)((n + 255) / 256), 256, 0, stream>>>(x, emb, x_emb);
  }

  const int projBlocks = (TBROWS / 16) * (GDIM / 64) / 8; // 2048

  if (useG) {
    input_proj_kernel<<<projBlocks, 256, 0, stream>>>(x_emb, EDIM, hWih0f, b0f, Gf);
    input_proj_kernel<<<projBlocks, 256, 0, stream>>>(x_emb, EDIM, hWih0b, b0b, Gb);
    lstm_rec_kernel<<<2, 256, REC_LDS_BYTES, stream>>>(Gf, Gb, hWhh0f, hWhh0b, h1);
    input_proj_kernel<<<projBlocks, 256, 0, stream>>>(h1, 2 * HDIM, hWih1f, b1f, Gf);
    input_proj_kernel<<<projBlocks, 256, 0, stream>>>(h1, 2 * HDIM, hWih1b, b1b, Gb);
    lstm_rec_kernel<<<2, 256, REC_LDS_BYTES, stream>>>(Gf, Gb, hWhh1f, hWhh1b, h2);
  } else {
    lstm_layer_fused_kernel<<<2, 256, 0, stream>>>(x_emb, EDIM,
                                                   hWih0f, hWhh0f, b0f,
                                                   hWih0b, hWhh0b, b0b, h1);
    lstm_layer_fused_kernel<<<2, 256, 0, stream>>>(h1, 2 * HDIM,
                                                   hWih1f, hWhh1f, b1f,
                                                   hWih1b, hWhh1b, b1b, h2);
  }

  emission_kernel<<<TBROWS / 8, 256, 0, stream>>>(h2, W_out, b_out, em);
  crf_kernel<<<1, 32, 0, stream>>>(em, labels, mask, trans, startt, endt,
                                   (float*)d_out);
}
